// REFILAgent_14757507629755
// MI455X (gfx1250) — compile-verified, matrix operand order checked by reading, not run on hardware
//
#include <hip/hip_runtime.h>
#include <math.h>

typedef __attribute__((ext_vector_type(16))) _Float16 v16h;
typedef __attribute__((ext_vector_type(8)))  float    v8f;
typedef int v4i_vec __attribute__((vector_size(16)));

#define DEVINL __device__ __forceinline__
#define GAS __attribute__((address_space(1)))
#define LAS __attribute__((address_space(3)))

#if __has_builtin(__builtin_amdgcn_global_load_async_to_lds_b128) && \
    __has_builtin(__builtin_amdgcn_s_wait_asynccnt)
#define USE_ASYNC_LDS 1
#else
#define USE_ASYNC_LDS 0
#endif

constexpr int BS = 32, T = 64, NE = 32, NA = 8, ED = 128;
constexpr int H = 512, AD = 512, NH = 8, HD = 64, NACT = 32;
constexpr int R = BS * NA; // 256 GRU rows

// ---- WMMA fragment helpers (16x16x32 f16, wave32) -------------------------
// A (16x32 row-major) / B^T (16x32 row-major == B col-major): lane L holds
// row (L&15); K indices {0..7,16..23} for lanes 0-15, {8..15,24..31} for 16-31.
DEVINL v16h frag_ld(const _Float16* base, int stride) {
  const int lane = threadIdx.x & 31;
  const _Float16* p = base + (lane & 15) * stride + ((lane >> 4) << 3);
  v16h r;
#pragma unroll
  for (int i = 0; i < 8; ++i) r[i] = p[i];
#pragma unroll
  for (int i = 0; i < 8; ++i) r[i + 8] = p[16 + i];
  return r;
}

DEVINL v8f wmma_f16(v16h a, v16h b, v8f c) {
  return __builtin_amdgcn_wmma_f32_16x16x32_f16(
      /*neg_a=*/false, a, /*neg_b=*/false, b,
      /*c_mod=*/(short)0, c, /*reuse_a=*/false, /*reuse_b=*/false);
}

// ---- weight conversion ----------------------------------------------------
__global__ void k_convT(const float* __restrict__ src, _Float16* __restrict__ dst,
                        int K, int N) { // src [K][N] -> dst [N][K]
  int idx = blockIdx.x * 256 + threadIdx.x;
  if (idx < K * N) {
    int n = idx / K, k = idx % K;
    dst[idx] = (_Float16)src[k * N + n];
  }
}
__global__ void k_conv(const float* __restrict__ src, _Float16* __restrict__ dst, int n) {
  int idx = blockIdx.x * 256 + threadIdx.x;
  if (idx < n) dst[idx] = (_Float16)src[idx];
}

// ---- fused fc1 + QKV + attention per (b,t) --------------------------------
DEVINL void gemm_small(const _Float16* Alds, int kdim,
                       const _Float16* __restrict__ Bt, const float* __restrict__ bias,
                       _Float16* out, int mtiles, bool dorelu) {
  const int w = threadIdx.x >> 5, lane = threadIdx.x & 31;
  for (int ni = 0; ni < 2; ++ni) {
    const int col0 = w * 64 + ni * 32;
    for (int mt = 0; mt < mtiles; ++mt) {
      v8f acc0 = {}, acc1 = {};
      for (int kk = 0; kk < kdim; kk += 32) {
        v16h a  = frag_ld(Alds + (mt * 16) * kdim + kk, kdim);
        v16h b0 = frag_ld(Bt + (size_t)col0 * kdim + kk, kdim);
        v16h b1 = frag_ld(Bt + (size_t)(col0 + 16) * kdim + kk, kdim);
        acc0 = wmma_f16(a, b0, acc0);
        acc1 = wmma_f16(a, b1, acc1);
      }
      const int m0 = mt * 16 + ((lane >> 4) << 3);
      const int nl = lane & 15;
      const float bv0 = bias[col0 + nl], bv1 = bias[col0 + 16 + nl];
#pragma unroll
      for (int i = 0; i < 8; ++i) {
        float v0 = acc0[i] + bv0;
        float v1 = acc1[i] + bv1;
        if (dorelu) { v0 = fmaxf(v0, 0.f); v1 = fmaxf(v1, 0.f); }
        out[(m0 + i) * AD + col0 + nl]      = (_Float16)v0;
        out[(m0 + i) * AD + col0 + nl + 16] = (_Float16)v1;
      }
    }
  }
}

__global__ __launch_bounds__(256)
void k_attention(const float* __restrict__ inputs, const int* __restrict__ obs_mask,
                 const int* __restrict__ scen,
                 const _Float16* __restrict__ W1t, const float* __restrict__ b1,
                 const _Float16* __restrict__ Wqt, const float* __restrict__ bq,
                 const _Float16* __restrict__ Wkt, const float* __restrict__ bk,
                 const _Float16* __restrict__ Wvt, const float* __restrict__ bv,
                 _Float16* __restrict__ att_x) {
  __shared__ _Float16 xin[NE * ED];      //  8 KB
  __shared__ _Float16 xbuf[NE * H];      // 32 KB
  __shared__ _Float16 kvbuf[NE * AD];    // 32 KB (K, then reused for V)
  __shared__ _Float16 qbuf[16 * AD];     // 16 KB (rows 0..7 valid)
  __shared__ float    wsc[NH * NA * NE]; //  8 KB => 96 KB total

  const int bt = blockIdx.x; // b*T + t
  const int b = bt >> 6, t = bt & 63;
  const int tid = threadIdx.x;

  // a) stage inputs (f32 -> f16)
  const float* inp = inputs + (size_t)bt * NE * ED;
  for (int i = tid; i < NE * ED; i += 256) xin[i] = (_Float16)inp[i];
  __syncthreads();

  // b) x = relu(inputs @ W1 + b1)
  gemm_small(xin, ED, W1t, b1, xbuf, 2, true);
  __syncthreads();

  // c) q (rows 0..15, only 8 used), k
  gemm_small(xbuf, H, Wqt, bq, qbuf, 1, false);
  gemm_small(xbuf, H, Wkt, bk, kvbuf, 2, false);
  __syncthreads();

  // d) scores + masked softmax; wave w == head, lane == entity
  {
    const int w = tid >> 5, e = tid & 31;
    float sc[NA];
#pragma unroll
    for (int a = 0; a < NA; ++a) {
      float s = 0.f;
      const _Float16* qp = qbuf + a * AD + w * HD;
      const _Float16* kp = kvbuf + e * AD + w * HD;
#pragma unroll 8
      for (int d = 0; d < HD; ++d) s += (float)qp[d] * (float)kp[d];
      s *= 0.125f; // 1/sqrt(64)
      if (obs_mask[(bt * NE + a) * NE + e]) s = -INFINITY;
      sc[a] = s;
    }
#pragma unroll
    for (int a = 0; a < NA; ++a) {
      float mx = sc[a];
      for (int off = 16; off > 0; off >>= 1) mx = fmaxf(mx, __shfl_xor(mx, off, 32));
      float ex = (mx == -INFINITY || sc[a] == -INFINITY) ? 0.f : expf(sc[a] - mx);
      float sm = ex;
      for (int off = 16; off > 0; off >>= 1) sm += __shfl_xor(sm, off, 32);
      wsc[(w * NA + a) * NE + e] = (sm > 0.f) ? ex / sm : 0.f; // NaN rows -> 0
    }
  }
  __syncthreads();

  // e) v = relu(x @ Wv + bv), overwrite kvbuf
  gemm_small(xbuf, H, Wvt, bv, kvbuf, 2, true);
  __syncthreads();

  // f) att = w @ v, agent mask, store f16 in xs layout [t][b*NA+a][c]
  for (int i = tid; i < NA * AD; i += 256) {
    const int a = i >> 9, c = i & 511, hh = c >> 6;
    float s = 0.f;
    const float* wp = wsc + (hh * NA + a) * NE;
#pragma unroll 8
    for (int e = 0; e < NE; ++e) s += wp[e] * (float)kvbuf[e * AD + c];
    if (scen[bt * NE + a]) s = 0.f;
    att_x[((size_t)t * R + b * NA + a) * AD + c] = (_Float16)s;
  }
}

// ---- GRU step t: fused gi + gh GEMMs + gates ------------------------------
// grid (8,4): 32-row x 128-col block tiles; 8 waves, each a 32x16 tile
// (2 m-subtiles sharing each B fragment).
__global__ __launch_bounds__(256)
void k_gru(int t, const _Float16* __restrict__ att_x,
           const float* __restrict__ hprev, float* __restrict__ hnext,
           const _Float16* __restrict__ Wih16, const float* __restrict__ bih,
           const _Float16* __restrict__ Whh16, const float* __restrict__ bhh,
           const int* __restrict__ scen,
           _Float16* __restrict__ hs16, float* __restrict__ out_hs) {
  __shared__ __align__(16) _Float16 atile[32 * 32]; // 2 KB
  __shared__ __align__(16) _Float16 htile[32 * 32]; // 2 KB
  const int tid = threadIdx.x;
  const int w = tid >> 5, lane = tid & 31;
  const int row0 = blockIdx.x * 32;
  const int col0 = blockIdx.y * 128;

  v8f sr[2] = {}, sz[2] = {}, gi[2] = {}, gh[2] = {};

  for (int kk = 0; kk < H; kk += 32) {
#if USE_ASYNC_LDS
    // 32x32 f16 tile = 2KB = 128 lanes x 16B: async global->LDS, no VGPR hop
    if (tid < 128) {
      const _Float16* gp =
          att_x + ((size_t)t * R + row0 + (tid >> 2)) * AD + kk + (tid & 3) * 8;
      __builtin_amdgcn_global_load_async_to_lds_b128(
          (GAS v4i_vec*)gp, (LAS v4i_vec*)(atile + tid * 8), 0, 0);
    }
    for (int i = tid; i < 32 * 32; i += 256) {
      const int r = i >> 5, c = i & 31;
      htile[i] = (_Float16)hprev[(size_t)(row0 + r) * H + kk + c];
    }
    __builtin_amdgcn_s_wait_asynccnt(0);
    __syncthreads();
#else
    for (int i = tid; i < 32 * 32; i += 256) {
      const int r = i >> 5, c = i & 31;
      atile[i] = att_x[((size_t)t * R + row0 + r) * AD + kk + c];
      htile[i] = (_Float16)hprev[(size_t)(row0 + r) * H + kk + c];
    }
    __syncthreads();
#endif

    v16h aA0 = frag_ld(atile, 32);
    v16h aA1 = frag_ld(atile + 16 * 32, 32);
    v16h aH0 = frag_ld(htile, 32);
    v16h aH1 = frag_ld(htile + 16 * 32, 32);

    const size_t gc = col0 + w * 16;
    v16h bir = frag_ld(Wih16 + gc * H + kk, H);
    v16h bhr = frag_ld(Whh16 + gc * H + kk, H);
    v16h biz = frag_ld(Wih16 + (gc + H) * H + kk, H);
    v16h bhz = frag_ld(Whh16 + (gc + H) * H + kk, H);
    v16h bin = frag_ld(Wih16 + (gc + 2 * H) * H + kk, H);
    v16h bhn = frag_ld(Whh16 + (gc + 2 * H) * H + kk, H);

    sr[0] = wmma_f16(aA0, bir, sr[0]);
    sr[0] = wmma_f16(aH0, bhr, sr[0]);
    sr[1] = wmma_f16(aA1, bir, sr[1]);
    sr[1] = wmma_f16(aH1, bhr, sr[1]);
    sz[0] = wmma_f16(aA0, biz, sz[0]);
    sz[0] = wmma_f16(aH0, bhz, sz[0]);
    sz[1] = wmma_f16(aA1, biz, sz[1]);
    sz[1] = wmma_f16(aH1, bhz, sz[1]);
    gi[0] = wmma_f16(aA0, bin, gi[0]);
    gi[1] = wmma_f16(aA1, bin, gi[1]);
    gh[0] = wmma_f16(aH0, bhn, gh[0]);
    gh[1] = wmma_f16(aH1, bhn, gh[1]);
    __syncthreads();
  }

  const int melem = (lane >> 4) << 3;
  const int nl = lane & 15;
  const int col = col0 + w * 16 + nl;
  const float br = bih[col] + bhh[col];
  const float bz = bih[col + H] + bhh[col + H];
  const float bin_b = bih[col + 2 * H];
  const float bhn_b = bhh[col + 2 * H];
#pragma unroll
  for (int ms = 0; ms < 2; ++ms) {
#pragma unroll
    for (int i = 0; i < 8; ++i) {
      const int row = row0 + ms * 16 + melem + i;
      const float rr = 1.f / (1.f + expf(-(sr[ms][i] + br)));
      const float zz = 1.f / (1.f + expf(-(sz[ms][i] + bz)));
      const float nn = tanhf(gi[ms][i] + bin_b + rr * (gh[ms][i] + bhn_b));
      const float hp = hprev[(size_t)row * H + col];
      const float hv = (1.f - zz) * nn + zz * hp;
      hnext[(size_t)row * H + col] = hv;
      hs16[((size_t)t * R + row) * H + col] = (_Float16)hv;
      const int b = row >> 3, a = row & 7;
      out_hs[(((size_t)b * T + t) * NA + a) * H + col] =
          scen[(b * T + t) * NE + a] ? 0.f : hv;
    }
  }
}

// ---- output head: qout = hs @ Wo + bo, masked -----------------------------
__global__ __launch_bounds__(256)
void k_qout(const _Float16* __restrict__ hs16, const _Float16* __restrict__ Wot,
            const float* __restrict__ bo, const int* __restrict__ scen,
            float* __restrict__ out_q) {
  const int tid = threadIdx.x;
  const int w = tid >> 5, lane = tid & 31;
  const int grow0 = blockIdx.x * 256 + w * 32;
  v8f acc[2][2] = {};
  for (int kk = 0; kk < H; kk += 32) {
    v16h a0 = frag_ld(hs16 + (size_t)grow0 * H + kk, H);
    v16h a1 = frag_ld(hs16 + ((size_t)grow0 + 16) * H + kk, H);
    v16h b0 = frag_ld(Wot + kk, H);
    v16h b1 = frag_ld(Wot + (size_t)16 * H + kk, H);
    acc[0][0] = wmma_f16(a0, b0, acc[0][0]);
    acc[0][1] = wmma_f16(a0, b1, acc[0][1]);
    acc[1][0] = wmma_f16(a1, b0, acc[1][0]);
    acc[1][1] = wmma_f16(a1, b1, acc[1][1]);
  }
  const int melem = (lane >> 4) << 3, nl = lane & 15;
#pragma unroll
  for (int ms = 0; ms < 2; ++ms)
#pragma unroll
    for (int ns = 0; ns < 2; ++ns)
#pragma unroll
      for (int i = 0; i < 8; ++i) {
        const int gr = grow0 + ms * 16 + melem + i;
        const int n = ns * 16 + nl;
        const int t = gr >> 8, r = gr & 255;
        const int b = r >> 3, a = r & 7;
        float v = acc[ms][ns][i] + bo[n];
        if (scen[(b * T + t) * NE + a]) v = 0.f;
        out_q[(((size_t)b * T + t) * NA + a) * NACT + n] = v;
      }
}

// ---- host launch ----------------------------------------------------------
extern "C" void kernel_launch(void* const* d_in, const int* in_sizes, int n_in,
                              void* d_out, int out_size, void* d_ws, size_t ws_size,
                              hipStream_t stream) {
  (void)in_sizes; (void)n_in; (void)out_size; (void)ws_size;
  const float* inputs = (const float*)d_in[0];
  const int*   obs    = (const int*)d_in[1];
  const int*   scen   = (const int*)d_in[2];
  const float* h0     = (const float*)d_in[3];
  const float* W1 = (const float*)d_in[4];  const float* b1 = (const float*)d_in[5];
  const float* Wq = (const float*)d_in[6];  const float* bq = (const float*)d_in[7];
  const float* Wk = (const float*)d_in[8];  const float* bk = (const float*)d_in[9];
  const float* Wv = (const float*)d_in[10]; const float* bv = (const float*)d_in[11];
  const float* Wih = (const float*)d_in[12]; const float* bih = (const float*)d_in[13];
  const float* Whh = (const float*)d_in[14]; const float* bhh = (const float*)d_in[15];
  const float* Wo = (const float*)d_in[16]; const float* bo = (const float*)d_in[17];

  char* ws = (char*)d_ws;
  size_t off = 0;
  auto take = [&](size_t bytes) { char* p = ws + off; off += (bytes + 255) & ~(size_t)255; return p; };
  _Float16* W1t   = (_Float16*)take((size_t)H * ED * 2);
  _Float16* Wqt   = (_Float16*)take((size_t)AD * H * 2);
  _Float16* Wkt   = (_Float16*)take((size_t)AD * H * 2);
  _Float16* Wvt   = (_Float16*)take((size_t)AD * H * 2);
  _Float16* Wih16 = (_Float16*)take((size_t)3 * H * AD * 2);
  _Float16* Whh16 = (_Float16*)take((size_t)3 * H * H * 2);
  _Float16* Wot   = (_Float16*)take((size_t)NACT * H * 2);
  _Float16* att_x = (_Float16*)take((size_t)T * R * AD * 2);
  _Float16* hs16  = (_Float16*)take((size_t)T * R * H * 2);
  float*    hbuf  = (float*)take((size_t)2 * R * H * 4);

  float* out_q  = (float*)d_out;                      // [BS,T,NA,NACT]
  float* out_hs = out_q + (size_t)BS * T * NA * NACT; // [BS,T,NA,H]

  // weight conversion (B^T layouts for WMMA)
  k_convT<<<(H * ED + 255) / 256, 256, 0, stream>>>(W1, W1t, ED, H);
  k_convT<<<(AD * H + 255) / 256, 256, 0, stream>>>(Wq, Wqt, H, AD);
  k_convT<<<(AD * H + 255) / 256, 256, 0, stream>>>(Wk, Wkt, H, AD);
  k_convT<<<(AD * H + 255) / 256, 256, 0, stream>>>(Wv, Wvt, H, AD);
  k_conv <<<(3 * H * AD + 255) / 256, 256, 0, stream>>>(Wih, Wih16, 3 * H * AD);
  k_conv <<<(3 * H * H + 255) / 256, 256, 0, stream>>>(Whh, Whh16, 3 * H * H);
  k_convT<<<(NACT * H + 255) / 256, 256, 0, stream>>>(Wo, Wot, H, NACT);

  // fused fc1 + QKV + attention
  k_attention<<<BS * T, 256, 0, stream>>>(inputs, obs, scen,
                                          W1t, b1, Wqt, bq, Wkt, bk, Wvt, bv, att_x);

  // sequential GRU over T
  for (int t = 0; t < T; ++t) {
    const float* hp = (t == 0) ? h0 : hbuf + (size_t)((t - 1) & 1) * R * H;
    float* hn = hbuf + (size_t)(t & 1) * R * H;
    k_gru<<<dim3(8, 4), 256, 0, stream>>>(t, att_x, hp, hn, Wih16, bih, Whh16, bhh,
                                          scen, hs16, out_hs);
  }

  // output head
  k_qout<<<(T * R) / 256, 256, 0, stream>>>(hs16, Wot, bo, scen, out_q);
}
